// MoE_19172734010056
// MI455X (gfx1250) — matrix operand order repaired and checked by reading, not compiled, not run on hardware
//
#include <hip/hip_runtime.h>
#include <hip/hip_bf16.h>
#include <math.h>

#define D_MODEL 1024
#define D_FF    4096
#define N_EXP   8
#define N_TOK   8192
#define OUT_Y   (N_TOK * D_MODEL)
#define W_ELEMS ((size_t)N_EXP * D_FF * D_MODEL)   // per weight tensor

typedef __attribute__((ext_vector_type(16))) __bf16          v16bf;
typedef __attribute__((ext_vector_type(8)))  __bf16          v8bf;
typedef __attribute__((ext_vector_type(8)))  float           v8f;
typedef __attribute__((ext_vector_type(4)))  float           v4f;
typedef __attribute__((ext_vector_type(4)))  __bf16          v4bf;
typedef __attribute__((ext_vector_type(8)))  unsigned short  v8us;
typedef __attribute__((ext_vector_type(16))) unsigned short  v16us;
typedef __attribute__((ext_vector_type(4)))  unsigned short  v4us;

// hardware fp32 -> bf16 (fptrunc; backend selects native cvt on gfx1250)
__device__ __forceinline__ unsigned short f32_bf16(float f) {
    return __builtin_bit_cast(unsigned short, (__bf16)f);
}

// 16 contiguous fp32 -> v16bf (for the no-preconvert fallback path)
__device__ __forceinline__ v16bf cvtB(const float* p) {
    v8f a = *(const v8f*)p;
    v8f b = *(const v8f*)(p + 8);
    v8bf ab = __builtin_convertvector(a, v8bf);
    v8bf bb = __builtin_convertvector(b, v8bf);
    return __builtin_shufflevector(ab, bb, 0,1,2,3,4,5,6,7,8,9,10,11,12,13,14,15);
}

// ---------------------------------------------------------------- zero
__global__ __launch_bounds__(256) void zero_k(float* __restrict__ out,
                                              int* __restrict__ smallws) {
    int i = blockIdx.x * 256 + threadIdx.x;
    if (i < OUT_Y) out[i] = 0.0f;
    if (i < 17) smallws[i] = 0;   // importance[8] + zsum[1] + counts[8]
}

// ---------------------------------------------------------------- weight preconvert
__global__ __launch_bounds__(256) void convert_w_k(
    const float* __restrict__ w1, const float* __restrict__ w2,
    unsigned short* __restrict__ w1b, unsigned short* __restrict__ w2b)
{
    const size_t nchunks = W_ELEMS / 8;                 // 8 elems per thread
    size_t i = (size_t)blockIdx.x * 256 + threadIdx.x;
    const float* src;
    unsigned short* dst;
    if (i < nchunks)      { src = w1; dst = w1b; }
    else if (i < 2 * nchunks) { src = w2; dst = w2b; i -= nchunks; }
    else return;
    const size_t off = i * 8;
    v8f v = *(const v8f*)(src + off);
    v8bf b = __builtin_convertvector(v, v8bf);
    *(v8us*)(dst + off) = __builtin_bit_cast(v8us, b);
}

// ---------------------------------------------------------------- router
__global__ __launch_bounds__(256) void router_k(
    const float* __restrict__ x, const float* __restrict__ rw,
    float* __restrict__ combine, float* __restrict__ imp,
    float* __restrict__ zsum, int* __restrict__ counts, int* __restrict__ idx)
{
    const int gtid = blockIdx.x * 256 + threadIdx.x;
    const int n    = gtid >> 5;
    const int lane = gtid & 31;
    if (n >= N_TOK) return;

    float acc[N_EXP];
#pragma unroll
    for (int e = 0; e < N_EXP; ++e) acc[e] = 0.0f;

    const float* xr = x + (size_t)n * D_MODEL;
    for (int d = lane; d < D_MODEL; d += 32) {
        const float xv = xr[d];
#pragma unroll
        for (int e = 0; e < N_EXP; ++e) acc[e] += xv * rw[e * D_MODEL + d];
    }
#pragma unroll
    for (int e = 0; e < N_EXP; ++e) {
#pragma unroll
        for (int off = 16; off > 0; off >>= 1)
            acc[e] += __shfl_xor(acc[e], off);
    }
    if (lane != 0) return;

    float mx = acc[0];
#pragma unroll
    for (int e = 1; e < N_EXP; ++e) mx = fmaxf(mx, acc[e]);
    float p[N_EXP], s = 0.0f;
#pragma unroll
    for (int e = 0; e < N_EXP; ++e) { p[e] = __expf(acc[e] - mx); s += p[e]; }
    const float lse = mx + logf(s);
    const float inv = 1.0f / s;
#pragma unroll
    for (int e = 0; e < N_EXP; ++e) p[e] *= inv;

    int i1 = 0; float v1 = p[0];
#pragma unroll
    for (int e = 1; e < N_EXP; ++e) if (p[e] > v1) { v1 = p[e]; i1 = e; }
    int i2 = (i1 == 0) ? 1 : 0; float v2 = p[i2];
#pragma unroll
    for (int e = 0; e < N_EXP; ++e)
        if (e != i1 && p[e] > v2) { v2 = p[e]; i2 = e; }

    const float tsum = v1 + v2 + 1e-9f;
    const float t1 = v1 / tsum, t2 = v2 / tsum;
#pragma unroll
    for (int e = 0; e < N_EXP; ++e)
        combine[(size_t)n * N_EXP + e] = (e == i1) ? t1 : ((e == i2) ? t2 : 0.0f);
#pragma unroll
    for (int e = 0; e < N_EXP; ++e) atomicAdd(&imp[e], p[e]);
    atomicAdd(zsum, lse * lse);

    int s1 = atomicAdd(&counts[i1], 1); idx[i1 * N_TOK + s1] = n;
    int s2 = atomicAdd(&counts[i2], 1); idx[i2 * N_TOK + s2] = n;
}

// ---------------------------------------------------------------- aux losses
__global__ void finalize_k(const float* __restrict__ imp,
                           const float* __restrict__ zsum,
                           const int* __restrict__ counts,
                           float* __restrict__ out)
{
    if (threadIdx.x != 0 || blockIdx.x != 0) return;
    float impsum = 0.0f, loadsum = 0.0f;
    for (int e = 0; e < N_EXP; ++e) { impsum += imp[e]; loadsum += (float)counts[e]; }
    float bal = 0.0f;
    for (int e = 0; e < N_EXP; ++e)
        bal += (imp[e] / (impsum + 1e-9f)) * ((float)counts[e] / (loadsum + 1e-9f));
    bal *= (float)N_EXP;
    const float z = zsum[0] / (float)N_TOK;
    out[OUT_Y] = 0.01f * bal + 0.001f * z;
    for (int e = 0; e < N_EXP; ++e) out[OUT_Y + 1 + e] = (float)counts[e];
    for (int e = 0; e < N_EXP; ++e) out[OUT_Y + 9 + e] = imp[e];
}

// ---------------------------------------------------------------- expert MLP (WMMA)
// TM = 32 tokens/block, 8 waves. f-chunk = 64: each wave computes one 16x16 h
// tile (K=1024, 32 WMMAs), then owns a 32x128 output slab (K=64, 32 WMMAs).
// PRE=true: weights pre-converted to bf16 in workspace (pure vmem->wmma loop).
template <bool PRE>
__global__ __launch_bounds__(256) void moe_mlp(
    const float* __restrict__ x,
    const float* __restrict__ w1,  const float* __restrict__ b1,
    const float* __restrict__ w2,  const float* __restrict__ b2,
    const unsigned short* __restrict__ w1bf, const unsigned short* __restrict__ w2bf,
    const float* __restrict__ combine,
    const int* __restrict__ counts, const int* __restrict__ idx,
    float* __restrict__ y)
{
    const int e     = blockIdx.y;
    const int tile  = blockIdx.x;
    const int count = counts[e];
    const int base  = tile * 32;
    if (base >= count) return;           // routed-sparse: ~3/4 of blocks exit here

    __shared__ unsigned short xlds[32 * D_MODEL];  // 64 KB bf16 token tile
    __shared__ unsigned short hlds[32 * 64];       // 4 KB bf16 hidden chunk
    __shared__ float cw[32];
    __shared__ int   toks[32];

    const int tid = threadIdx.x;
    if (tid < 32) {
        const int valid = (base + tid) < count;
        const int t = valid ? idx[e * N_TOK + base + tid] : 0;
        toks[tid] = t;
        cw[tid]   = valid ? combine[(size_t)t * N_EXP + e] : 0.0f;
    }
    __syncthreads();

    // x tile -> LDS bf16 (float4 loads, 4x bf16 stores)
    for (int c4 = tid; c4 < 32 * (D_MODEL / 4); c4 += 256) {
        const int r = c4 >> 8;                 // 256 float4 per row
        const int c = (c4 & 255) * 4;
        v4f v = *(const v4f*)(x + (size_t)toks[r] * D_MODEL + c);
        v4bf b = __builtin_convertvector(v, v4bf);
        *(v4us*)&xlds[r * D_MODEL + c] = __builtin_bit_cast(v4us, b);
    }
    __syncthreads();

    const int lane = tid & 31;
    const int wave = tid >> 5;
    const int col  = lane & 15;
    const int hi   = lane >> 4;
    const int r1   = wave & 1;      // GEMM1: token row-tile of this wave
    const int fct  = wave >> 1;     // GEMM1: f column-tile of this wave (0..3)

    const float* w1e = w1 + (size_t)e * D_FF * D_MODEL;
    const float* w2e = w2 + (size_t)e * D_MODEL * D_FF;
    const unsigned short* w1be = w1bf + (size_t)e * D_FF * D_MODEL;
    const unsigned short* w2be = w2bf + (size_t)e * D_MODEL * D_FF;
    const float* b1e = b1 + (size_t)e * D_FF;
    const float* b2e = b2 + (size_t)e * D_MODEL;

    v8f acc[2][8];
#pragma unroll
    for (int r = 0; r < 2; ++r)
#pragma unroll
        for (int t = 0; t < 8; ++t) acc[r][t] = v8f{};

    for (int f0 = 0; f0 < D_FF; f0 += 64) {
        // ---- GEMM1: one 16x16 h tile per wave over K = D_MODEL ----
        const int frow = f0 + fct * 16 + col;
        if (PRE && f0 + 64 < D_FF)
            __builtin_prefetch(w1be + (size_t)(frow + 64) * D_MODEL, 0, 0);

        v8f hacc = v8f{};
#pragma unroll 4
        for (int k0 = 0; k0 < D_MODEL; k0 += 32) {
            // A (16x32 bf16) from LDS: lanes 0-15 K{0..15}; 16-31 K{16..31}
            const int arow = (r1 * 16 + col) * D_MODEL;
            const v8us alo = *(const v8us*)&xlds[arow + k0 + hi * 16];
            const v8us ahi = *(const v8us*)&xlds[arow + k0 + hi * 16 + 8];
            const v16us au = __builtin_shufflevector(alo, ahi,
                0,1,2,3,4,5,6,7,8,9,10,11,12,13,14,15);
            // B (32x16 bf16): lane = w1 row (output f), K 0-15 / 16-31 by half-wave
            v16bf bf;
            if (PRE) {
                bf = __builtin_bit_cast(v16bf,
                    *(const v16us*)(w1be + (size_t)frow * D_MODEL + k0 + hi * 16));
            } else {
                bf = cvtB(w1e + (size_t)frow * D_MODEL + k0 + hi * 16);
            }
            hacc = __builtin_amdgcn_wmma_f32_16x16x32_bf16(
                false, __builtin_bit_cast(v16bf, au), false, bf,
                (short)0, hacc, false, false);
        }
        // bias + exact GELU -> bf16 h chunk in LDS ([32 tokens][64 f])
        const float bias = b1e[frow];
#pragma unroll
        for (int j = 0; j < 8; ++j) {
            const float v = hacc[j] + bias;
            const float g = 0.5f * v * (1.0f + erff(v * 0.70710678118654752f));
            hlds[(r1 * 16 + j + hi * 8) * 64 + fct * 16 + col] = f32_bf16(g);
        }
        __syncthreads();

        // ---- GEMM2: 32x128 slab per wave over K = 64 ----
        v16us ahf[2][2];
#pragma unroll
        for (int r = 0; r < 2; ++r)
#pragma unroll
            for (int ks = 0; ks < 2; ++ks) {
                const int hrow = (r * 16 + col) * 64;
                const v8us lo  = *(const v8us*)&hlds[hrow + ks * 32 + hi * 16];
                const v8us hi8 = *(const v8us*)&hlds[hrow + ks * 32 + hi * 16 + 8];
                ahf[r][ks] = __builtin_shufflevector(lo, hi8,
                    0,1,2,3,4,5,6,7,8,9,10,11,12,13,14,15);
            }
#pragma unroll
        for (int t = 0; t < 8; ++t) {
            const int d = wave * 128 + t * 16 + col;
#pragma unroll
            for (int ks = 0; ks < 2; ++ks) {
                v16bf bf;
                if (PRE) {
                    bf = __builtin_bit_cast(v16bf,
                        *(const v16us*)(w2be + (size_t)d * D_FF + f0 + ks * 32 + hi * 16));
                } else {
                    bf = cvtB(w2e + (size_t)d * D_FF + f0 + ks * 32 + hi * 16);
                }
#pragma unroll
                for (int r = 0; r < 2; ++r) {
                    acc[r][t] = __builtin_amdgcn_wmma_f32_16x16x32_bf16(
                        false, __builtin_bit_cast(v16bf, ahf[r][ks]), false, bf,
                        (short)0, acc[r][t], false, false);
                }
            }
        }
        __syncthreads();
    }

    // (out + b2) * combine -> scatter-add into y (two experts per token)
#pragma unroll
    for (int t = 0; t < 8; ++t) {
        const int d = wave * 128 + t * 16 + col;
        const float bias2 = b2e[d];
#pragma unroll
        for (int r = 0; r < 2; ++r) {
#pragma unroll
            for (int j = 0; j < 8; ++j) {
                const int m = r * 16 + j + hi * 8;
                const float val = (acc[r][t][j] + bias2) * cw[m];
                atomicAdd(&y[(size_t)toks[m] * D_MODEL + d], val);
            }
        }
    }
}

// ---------------------------------------------------------------- launch
extern "C" void kernel_launch(void* const* d_in, const int* in_sizes, int n_in,
                              void* d_out, int out_size, void* d_ws, size_t ws_size,
                              hipStream_t stream) {
    const float* x  = (const float*)d_in[0];
    const float* rw = (const float*)d_in[1];
    const float* w1 = (const float*)d_in[2];
    const float* b1 = (const float*)d_in[3];
    const float* w2 = (const float*)d_in[4];
    const float* b2 = (const float*)d_in[5];
    float* out = (float*)d_out;

    // workspace: combine | imp | zsum | counts | idx | (aligned) w1b | w2b
    unsigned char* wsb = (unsigned char*)d_ws;
    float* combine = (float*)wsb;                         // N_TOK*8 floats
    float* imp     = combine + (size_t)N_TOK * N_EXP;     // 8
    float* zsum    = imp + N_EXP;                         // 1
    int*   counts  = (int*)(zsum + 1);                    // 8
    int*   idx     = counts + N_EXP;                      // N_TOK*8 ints
    size_t fixed   = (size_t)N_TOK * N_EXP * 4 + 17 * 4 + (size_t)N_TOK * N_EXP * 4;
    fixed = (fixed + 255) & ~(size_t)255;
    unsigned short* w1b = (unsigned short*)(wsb + fixed);
    unsigned short* w2b = w1b + W_ELEMS;
    const size_t need = fixed + 2 * W_ELEMS * sizeof(unsigned short);
    const bool pre = (ws_size >= need);

    zero_k<<<(OUT_Y + 255) / 256, 256, 0, stream>>>(out, (int*)imp);
    if (pre) {
        const size_t threads = 2 * (W_ELEMS / 8);
        convert_w_k<<<(unsigned)((threads + 255) / 256), 256, 0, stream>>>(w1, w2, w1b, w2b);
    }
    router_k<<<(N_TOK * 32 + 255) / 256, 256, 0, stream>>>(
        x, rw, combine, imp, zsum, counts, idx);
    finalize_k<<<1, 32, 0, stream>>>(imp, zsum, counts, out);

    dim3 grid(N_TOK / 32, N_EXP);
    if (pre) {
        moe_mlp<true><<<grid, 256, 0, stream>>>(
            x, w1, b1, w2, b2, w1b, w2b, combine, counts, idx, out);
    } else {
        moe_mlp<false><<<grid, 256, 0, stream>>>(
            x, w1, b1, w2, b2, w1b, w2b, combine, counts, idx, out);
    }
}
